// WaveFunctionCollapseDecoder_11407433138223
// MI455X (gfx1250) — matrix operand order repaired and checked by reading, not compiled, not run on hardware
//
#include <hip/hip_runtime.h>
#include <hip/hip_bf16.h>

// ---------------------------------------------------------------------------
// Types for CDNA5 WMMA (wave32): 16x16x32 bf16 -> f32
// ---------------------------------------------------------------------------
typedef __attribute__((ext_vector_type(16))) __bf16 v16bf;
typedef __attribute__((ext_vector_type(8)))  __bf16 v8bf;
typedef __attribute__((ext_vector_type(8)))  float  v8f;
typedef __attribute__((ext_vector_type(4)))  int    v4i;

#define GLOBAL_AS __attribute__((address_space(1)))
#define LDS_AS    __attribute__((address_space(3)))

#if defined(__has_builtin)
#if __has_builtin(__builtin_amdgcn_global_load_async_to_lds_b128)
#define HAVE_ASYNC_LDS 1
#endif
#endif
#ifndef HAVE_ASYNC_LDS
#define HAVE_ASYNC_LDS 0
#endif

__device__ __forceinline__ float gelu_f(float v) {
    // exact GELU: x * 0.5 * (1 + erf(x / sqrt(2)))
    return 0.5f * v * (1.0f + erff(v * 0.70710678118654752f));
}

template <int N>
__device__ __forceinline__ void wait_async() {
#if defined(__has_builtin) && __has_builtin(__builtin_amdgcn_s_wait_asynccnt)
    __builtin_amdgcn_s_wait_asynccnt((unsigned short)N);
#else
    asm volatile("s_wait_asynccnt %0" ::"i"(N) : "memory");
#endif
}

// ---------------------------------------------------------------------------
// LDS tile geometry: 128 rows x 32 cols bf16, padded row stride 40 elements
// (80B = 20 banks -> the 16-lane b128 fragment reads are bank-conflict-free)
// ---------------------------------------------------------------------------
constexpr int TILE_R    = 128;
constexpr int TILE_K    = 32;
constexpr int LDS_LD    = 40;   // elements
constexpr int TILE_ELEM = TILE_R * LDS_LD;

// Cooperative copy of a 128x32 bf16 tile (global, row-major ld=K) into LDS.
// 256 threads * 2 chunks * 16B = 8KB.
#if HAVE_ASYNC_LDS
__device__ __forceinline__ void tile_copy_async(const __bf16* __restrict__ g, int ld,
                                                __bf16* __restrict__ l) {
    const int c  = (threadIdx.x & 3) * 8;   // col chunk: 0,8,16,24
    const int r0 = threadIdx.x >> 2;        // 0..63
#pragma unroll
    for (int rr = 0; rr < 2; ++rr) {
        const int row = r0 + rr * 64;
        __builtin_amdgcn_global_load_async_to_lds_b128(
            (GLOBAL_AS v4i*)(g + (size_t)row * ld + c),
            (LDS_AS v4i*)(l + row * LDS_LD + c), 0, 0);
    }
}
#else
__device__ __forceinline__ void tile_load_regs(const __bf16* __restrict__ g, int ld,
                                               uint4 rg[2]) {
    const int c  = (threadIdx.x & 3) * 8;
    const int r0 = threadIdx.x >> 2;
#pragma unroll
    for (int rr = 0; rr < 2; ++rr)
        rg[rr] = *reinterpret_cast<const uint4*>(g + (size_t)(r0 + rr * 64) * ld + c);
}
__device__ __forceinline__ void tile_store_lds(__bf16* __restrict__ l, const uint4 rg[2]) {
    const int c  = (threadIdx.x & 3) * 8;
    const int r0 = threadIdx.x >> 2;
#pragma unroll
    for (int rr = 0; rr < 2; ++rr)
        *reinterpret_cast<uint4*>(l + (r0 + rr * 64) * LDS_LD + c) = rg[rr];
}
#endif

// A fragment (16x32 bf16, MxK) from LDS tile:
// lanes 0-15: row M=lane, K={0..7,16..23}; lanes 16-31: row M=lane-16, K={8..15,24..31}
__device__ __forceinline__ v16bf lds_frag_a(const __bf16* l, int row, int half) {
    const __bf16* p = l + row * LDS_LD + half * 8;
    v8bf lo = *reinterpret_cast<const v8bf*>(p);
    v8bf hi = *reinterpret_cast<const v8bf*>(p + 16);
    v16bf r;
#pragma unroll
    for (int i = 0; i < 8; ++i) { r[i] = lo[i]; r[i + 8] = hi[i]; }
    return r;
}

// B fragment (32x16 bf16, KxN) from LDS tile holding W rows (N-major):
// lanes 0-15: col N=lane, K={0..7,16..23}; lanes 16-31: col N=lane-16, K={8..15,24..31}
__device__ __forceinline__ v16bf lds_frag_b(const __bf16* l, int row, int half) {
    const __bf16* p = l + row * LDS_LD + half * 8;
    v8bf lo = *reinterpret_cast<const v8bf*>(p);
    v8bf hi = *reinterpret_cast<const v8bf*>(p + 16);
    v16bf r;
#pragma unroll
    for (int i = 0; i < 8; ++i) { r[i] = lo[i]; r[i + 8] = hi[i]; }
    return r;
}

// ---------------------------------------------------------------------------
// WMMA GEMM: out = epilogue(A[M,K](bf16) @ W[N,K]^T(bf16) + bias)
// Block 256 thr = 8 waves; block tile 128(M)x128(N)x32(K), LDS double-buffered.
// Waves arranged 4(M)x2(N); wave tile 32x64 -> 8 accumulators, 8 WMMA/K-step.
// MODE 0: GELU -> bf16 store        MODE 1: x[m,n] += (acc+bias)*cw[m]*0.3
// MODE 2: plain fp32 store, no bias
// ---------------------------------------------------------------------------
template <int MODE>
__global__ __launch_bounds__(256)
void gemm_bf16_wmma(const __bf16* __restrict__ A, const __bf16* __restrict__ W,
                    const float* __restrict__ bias,
                    __bf16* __restrict__ outbf, float* __restrict__ outf,
                    float* __restrict__ x, const float* __restrict__ cw,
                    int N, int K) {
    __shared__ __bf16 sA[2][TILE_ELEM];
    __shared__ __bf16 sB[2][TILE_ELEM];

    const int tid  = threadIdx.x;
    const int wave = tid >> 5;
    const int lane = tid & 31;
    const int half = lane >> 4;
    const int r    = lane & 15;
    const int wm   = wave & 3;   // 4 M-waves * 32 rows
    const int wn   = wave >> 2;  // 2 N-waves * 64 cols

    const size_t m0 = (size_t)blockIdx.x * 128;
    const size_t n0 = (size_t)blockIdx.y * 128;
    const __bf16* Ag = A + m0 * (size_t)K;
    const __bf16* Wg = W + n0 * (size_t)K;

    v8f acc[2][4] = {};

    const int nk = K / TILE_K;

#if HAVE_ASYNC_LDS
    tile_copy_async(Ag, K, sA[0]);
    tile_copy_async(Wg, K, sB[0]);
    for (int k = 0; k < nk; ++k) {
        const int cur = k & 1;
        if (k + 1 < nk) {
            tile_copy_async(Ag + (size_t)(k + 1) * TILE_K, K, sA[cur ^ 1]);
            tile_copy_async(Wg + (size_t)(k + 1) * TILE_K, K, sB[cur ^ 1]);
            wait_async<4>();   // async completes in order: current tile's 4 ops done
        } else {
            wait_async<0>();
        }
        __syncthreads();       // all waves' LDS writes for `cur` visible

        v16bf a0 = lds_frag_a(sA[cur], wm * 32 + r, half);
        v16bf a1 = lds_frag_a(sA[cur], wm * 32 + 16 + r, half);
#pragma unroll
        for (int j = 0; j < 4; ++j) {
            v16bf b = lds_frag_b(sB[cur], wn * 64 + j * 16 + r, half);
            acc[0][j] = __builtin_amdgcn_wmma_f32_16x16x32_bf16(
                false, a0, false, b, (short)0, acc[0][j], false, false);
            acc[1][j] = __builtin_amdgcn_wmma_f32_16x16x32_bf16(
                false, a1, false, b, (short)0, acc[1][j], false, false);
        }
        __syncthreads();       // reads done before buffer is overwritten
    }
#else
    uint4 ra[2], rb[2];
    tile_load_regs(Ag, K, ra);
    tile_load_regs(Wg, K, rb);
    for (int k = 0; k < nk; ++k) {
        tile_store_lds(sA[0], ra);
        tile_store_lds(sB[0], rb);
        __syncthreads();
        if (k + 1 < nk) {   // prefetch next tile into registers during compute
            tile_load_regs(Ag + (size_t)(k + 1) * TILE_K, K, ra);
            tile_load_regs(Wg + (size_t)(k + 1) * TILE_K, K, rb);
        }
        v16bf a0 = lds_frag_a(sA[0], wm * 32 + r, half);
        v16bf a1 = lds_frag_a(sA[0], wm * 32 + 16 + r, half);
#pragma unroll
        for (int j = 0; j < 4; ++j) {
            v16bf b = lds_frag_b(sB[0], wn * 64 + j * 16 + r, half);
            acc[0][j] = __builtin_amdgcn_wmma_f32_16x16x32_bf16(
                false, a0, false, b, (short)0, acc[0][j], false, false);
            acc[1][j] = __builtin_amdgcn_wmma_f32_16x16x32_bf16(
                false, a1, false, b, (short)0, acc[1][j], false, false);
        }
        __syncthreads();
    }
#endif

    // Epilogue. D layout: m = 8*(lane/16) + vgpr, n = lane&15 within each 16x16.
#pragma unroll
    for (int mi = 0; mi < 2; ++mi) {
#pragma unroll
        for (int j = 0; j < 4; ++j) {
#pragma unroll
            for (int i = 0; i < 8; ++i) {
                const size_t row = m0 + (size_t)(wm * 32 + mi * 16 + half * 8 + i);
                const size_t col = n0 + (size_t)(wn * 64 + j * 16 + r);
                float val = acc[mi][j][i];
                if (MODE == 0) {
                    val += bias[col];
                    outbf[row * (size_t)N + col] = (__bf16)gelu_f(val);
                } else if (MODE == 1) {
                    val += bias[col];
                    x[row * (size_t)N + col] += val * cw[row] * 0.3f;
                } else {
                    outf[row * (size_t)N + col] = val;
                }
            }
        }
    }
}

// ---------------------------------------------------------------------------
// fp32 -> bf16 conversion (weights), grid-stride
// ---------------------------------------------------------------------------
__global__ __launch_bounds__(256)
void f2bf_kernel(const float* __restrict__ in, __bf16* __restrict__ out, size_t n) {
    size_t i = (size_t)blockIdx.x * 256 + threadIdx.x;
    const size_t stride = (size_t)gridDim.x * 256;
    for (; i < n; i += stride) out[i] = (__bf16)in[i];
}

// ---------------------------------------------------------------------------
// c-path: c_pooled = mean(c_states, axis=0); MLP 512->1024(gelu)->512
// ---------------------------------------------------------------------------
__global__ __launch_bounds__(256)
void cpath_kernel(const float* __restrict__ c_states,
                  const float* __restrict__ w1, const float* __restrict__ b1,
                  const float* __restrict__ w2, const float* __restrict__ b2,
                  float* __restrict__ cfp, __bf16* __restrict__ cbf) {
    __shared__ float pooled[512];
    __shared__ float h[1024];
    const int tid = threadIdx.x;
    for (int i = tid; i < 512; i += 256)
        pooled[i] = 0.25f * (c_states[i] + c_states[512 + i] +
                             c_states[1024 + i] + c_states[1536 + i]);
    __syncthreads();
    for (int j = tid; j < 1024; j += 256) {
        const float* wr = w1 + (size_t)j * 512;
        float s = b1[j];
        for (int k = 0; k < 512; ++k) s += pooled[k] * wr[k];
        h[j] = gelu_f(s);
    }
    __syncthreads();
    for (int j = tid; j < 512; j += 256) {
        const float* wr = w2 + (size_t)j * 1024;
        float s = b2[j];
        for (int k = 0; k < 1024; ++k) s += h[k] * wr[k];
        cfp[j] = s;
        cbf[j] = (__bf16)s;
    }
}

// ---------------------------------------------------------------------------
// Embedding: x[b,t,:] = embed_w[tokens[b,t]] + pos_w[t]
// ---------------------------------------------------------------------------
__global__ __launch_bounds__(256)
void embed_kernel(const int* __restrict__ tokens, const float* __restrict__ embed_w,
                  const float* __restrict__ pos_w, float* __restrict__ x,
                  int T, int D) {
    const int m = blockIdx.x;
    const int t = m % T;
    const int tok = tokens[m];
    const float* ew = embed_w + (size_t)tok * D;
    const float* pw = pos_w + (size_t)t * D;
    float* xr = x + (size_t)m * D;
    for (int i = threadIdx.x; i < D; i += 256) xr[i] = ew[i] + pw[i];
}

// ---------------------------------------------------------------------------
// Pack ci = [x, roll(x,+1,T), roll(x,-1,T), c_constraint] -> bf16 [M, 4D]
// ---------------------------------------------------------------------------
__global__ __launch_bounds__(256)
void pack_ci_kernel(const float* __restrict__ x, const __bf16* __restrict__ cbf,
                    __bf16* __restrict__ ci, int T, int D) {
    const int m = blockIdx.x;
    const int b = m / T;
    const int t = m % T;
    const int tl = (t == 0) ? (T - 1) : (t - 1);
    const int tr = (t == T - 1) ? 0 : (t + 1);
    const float* xc = x + (size_t)m * D;
    const float* xl = x + ((size_t)b * T + tl) * D;
    const float* xr = x + ((size_t)b * T + tr) * D;
    __bf16* row = ci + (size_t)m * 4 * D;
    for (int i = threadIdx.x; i < D; i += 256) {
        row[i]         = (__bf16)xc[i];
        row[D + i]     = (__bf16)xl[i];
        row[2 * D + i] = (__bf16)xr[i];
        row[3 * D + i] = cbf[i];
    }
}

// ---------------------------------------------------------------------------
// Entropy MLP per row: 512 -> 64 (gelu) -> 1 ; one 64-thread block per row
// ---------------------------------------------------------------------------
__global__ __launch_bounds__(64)
void entropy_kernel(const float* __restrict__ x, const float* __restrict__ w1,
                    const float* __restrict__ b1, const float* __restrict__ w2,
                    const float* __restrict__ b2, float* __restrict__ ent) {
    __shared__ float sx[512];
    __shared__ float sv[64];
    const int m = blockIdx.x;
    const float* xr = x + (size_t)m * 512;
    for (int i = threadIdx.x; i < 512; i += 64) sx[i] = xr[i];
    __syncthreads();
    const int j = threadIdx.x;
    const float* wr = w1 + (size_t)j * 512;
    float s = b1[j];
    for (int k = 0; k < 512; ++k) s += sx[k] * wr[k];
    sv[j] = gelu_f(s) * w2[j];
    __syncthreads();
    for (int off = 32; off > 0; off >>= 1) {
        if (j < off) sv[j] += sv[j + off];
        __syncthreads();
    }
    if (j == 0) ent[m] = sv[0] + b2[0];
}

// ---------------------------------------------------------------------------
// Softmax over T per batch: cw = softmax(-ent / temp); temp computed on device
// ---------------------------------------------------------------------------
__global__ __launch_bounds__(256)
void softmax_kernel(const float* __restrict__ ent, float* __restrict__ cw,
                    const float* __restrict__ base_temp, float step_scale, int T) {
    const int b = blockIdx.x;
    const float* e = ent + (size_t)b * T;
    const float sp = log1pf(expf(base_temp[0]));   // softplus
    const float inv_temp = 1.0f / (sp * step_scale);
    __shared__ float red[256];
    float lmax = -INFINITY;
    for (int t = threadIdx.x; t < T; t += 256) lmax = fmaxf(lmax, -e[t] * inv_temp);
    red[threadIdx.x] = lmax;
    __syncthreads();
    for (int off = 128; off > 0; off >>= 1) {
        if (threadIdx.x < off) red[threadIdx.x] = fmaxf(red[threadIdx.x], red[threadIdx.x + off]);
        __syncthreads();
    }
    const float mx = red[0];
    __syncthreads();
    float lsum = 0.0f;
    for (int t = threadIdx.x; t < T; t += 256) lsum += expf(-e[t] * inv_temp - mx);
    red[threadIdx.x] = lsum;
    __syncthreads();
    for (int off = 128; off > 0; off >>= 1) {
        if (threadIdx.x < off) red[threadIdx.x] += red[threadIdx.x + off];
        __syncthreads();
    }
    const float inv = 1.0f / red[0];
    for (int t = threadIdx.x; t < T; t += 256)
        cw[(size_t)b * T + t] = expf(-e[t] * inv_temp - mx) * inv;
}

// ---------------------------------------------------------------------------
// LayerNorm over D + bf16 cast
// ---------------------------------------------------------------------------
__global__ __launch_bounds__(256)
void ln_kernel(const float* __restrict__ x, const float* __restrict__ g,
               const float* __restrict__ bta, __bf16* __restrict__ xn, int D) {
    const int m = blockIdx.x;
    const float* xr = x + (size_t)m * D;
    __shared__ float red[256];
    float s = 0.0f;
    for (int i = threadIdx.x; i < D; i += 256) s += xr[i];
    red[threadIdx.x] = s;
    __syncthreads();
    for (int off = 128; off > 0; off >>= 1) {
        if (threadIdx.x < off) red[threadIdx.x] += red[threadIdx.x + off];
        __syncthreads();
    }
    const float mu = red[0] / (float)D;
    __syncthreads();
    float v = 0.0f;
    for (int i = threadIdx.x; i < D; i += 256) { float d = xr[i] - mu; v += d * d; }
    red[threadIdx.x] = v;
    __syncthreads();
    for (int off = 128; off > 0; off >>= 1) {
        if (threadIdx.x < off) red[threadIdx.x] += red[threadIdx.x + off];
        __syncthreads();
    }
    const float rstd = rsqrtf(red[0] / (float)D + 1e-5f);
    for (int i = threadIdx.x; i < D; i += 256)
        xn[(size_t)m * D + i] = (__bf16)((xr[i] - mu) * rstd * g[i] + bta[i]);
}

// ---------------------------------------------------------------------------
// Host-side orchestration
// ---------------------------------------------------------------------------
extern "C" void kernel_launch(void* const* d_in, const int* in_sizes, int n_in,
                              void* d_out, int out_size, void* d_ws, size_t ws_size,
                              hipStream_t stream) {
    (void)in_sizes; (void)n_in; (void)out_size; (void)ws_size;
    const int B = 8, T = 1024, D = 512, V = 32000;
    const int M = B * T;                 // 8192 rows
    const int N_STEPS = 6;

    const int*   tokens    = (const int*)  d_in[0];
    const float* c_states  = (const float*)d_in[1];
    const float* embed_w   = (const float*)d_in[2];
    const float* pos_w     = (const float*)d_in[3];
    const float* cn_w1     = (const float*)d_in[4];
    const float* cn_b1     = (const float*)d_in[5];
    const float* cn_w2     = (const float*)d_in[6];
    const float* cn_b2     = (const float*)d_in[7];
    const float* ent_w1    = (const float*)d_in[8];
    const float* ent_b1    = (const float*)d_in[9];
    const float* ent_w2    = (const float*)d_in[10];
    const float* ent_b2    = (const float*)d_in[11];
    const float* ctc_w1    = (const float*)d_in[12];
    const float* ctc_b1    = (const float*)d_in[13];
    const float* ctc_w2    = (const float*)d_in[14];
    const float* ctc_b2    = (const float*)d_in[15];
    const float* base_temp = (const float*)d_in[16];
    const float* ln_g      = (const float*)d_in[17];
    const float* ln_b      = (const float*)d_in[18];
    const float* head_w    = (const float*)d_in[19];

    // workspace layout (256B aligned)
    char* ws = (char*)d_ws;
    size_t off = 0;
    auto alloc = [&](size_t bytes) -> char* {
        char* p = ws + off;
        off = (off + bytes + 255) & ~(size_t)255;
        return p;
    };
    float*  x       = (float*) alloc((size_t)M * D * 4);        // residual stream (fp32)
    __bf16* ci      = (__bf16*)alloc((size_t)M * 4 * D * 2);    // [M, 2048] bf16
    __bf16* h       = (__bf16*)alloc((size_t)M * 2 * D * 2);    // [M, 1024] bf16
    __bf16* w1bf    = (__bf16*)alloc((size_t)(2 * D) * (4 * D) * 2);
    __bf16* w2bf    = (__bf16*)alloc((size_t)D * (2 * D) * 2);
    __bf16* headbf  = (__bf16*)alloc((size_t)V * D * 2);
    __bf16* xn      = (__bf16*)alloc((size_t)M * D * 2);
    float*  cfp     = (float*) alloc((size_t)D * 4);
    __bf16* cbf     = (__bf16*)alloc((size_t)D * 2);
    float*  ent     = (float*) alloc((size_t)M * 4);
    float*  cw      = (float*) alloc((size_t)M * 4);

    // 1) weight conversions to bf16
    f2bf_kernel<<<4096, 256, 0, stream>>>(cn_w1, w1bf, (size_t)(2 * D) * (4 * D));
    f2bf_kernel<<<2048, 256, 0, stream>>>(cn_w2, w2bf, (size_t)D * (2 * D));
    f2bf_kernel<<<8192, 256, 0, stream>>>(head_w, headbf, (size_t)V * D);

    // 2) c-constraint path (tiny)
    cpath_kernel<<<1, 256, 0, stream>>>(c_states, ctc_w1, ctc_b1, ctc_w2, ctc_b2, cfp, cbf);

    // 3) embedding
    embed_kernel<<<M, 256, 0, stream>>>(tokens, embed_w, pos_w, x, T, D);

    // 4) refinement steps
    for (int step = 0; step < N_STEPS; ++step) {
        const float step_scale = 1.0f - ((float)step / (float)N_STEPS) * 0.8f;

        pack_ci_kernel<<<M, 256, 0, stream>>>(x, cbf, ci, T, D);
        entropy_kernel<<<M, 64, 0, stream>>>(x, ent_w1, ent_b1, ent_w2, ent_b2, ent);
        softmax_kernel<<<B, 256, 0, stream>>>(ent, cw, base_temp, step_scale, T);

        // h = gelu(ci @ cn_w1^T + b1)   [M, 1024]
        {
            dim3 grid(M / 128, (2 * D) / 128);
            gemm_bf16_wmma<0><<<grid, 256, 0, stream>>>(
                ci, w1bf, cn_b1, h, nullptr, nullptr, nullptr, 2 * D, 4 * D);
        }
        // x += (h @ cn_w2^T + b2) * cw * 0.3   [M, 512]
        {
            dim3 grid(M / 128, D / 128);
            gemm_bf16_wmma<1><<<grid, 256, 0, stream>>>(
                h, w2bf, cn_b2, nullptr, nullptr, x, cw, D, 2 * D);
        }
    }

    // 5) LayerNorm -> bf16
    ln_kernel<<<M, 256, 0, stream>>>(x, ln_g, ln_b, xn, D);

    // 6) LM head: out = xn @ head_w^T   [M, 32000] fp32
    {
        dim3 grid(M / 128, V / 128);
        gemm_bf16_wmma<2><<<grid, 256, 0, stream>>>(
            xn, headbf, nullptr, nullptr, (float*)d_out, nullptr, nullptr, V, D);
    }
}
